// CompletionNet_54013508714742
// MI455X (gfx1250) — compile-verified, matrix operand order checked by reading, not run on hardware
//
#include <hip/hip_runtime.h>
#include <hip/hip_bf16.h>
#include <math.h>

// ---------------------------------------------------------------------------
// CompletionNet (sparse-conv U-Net, dense-masked formulation) for MI455X.
//  * conv layers = implicit GEMM on V_WMMA_F32_16X16X32_BF16 (f32 accum)
//  * activations: bf16 channels-last [D,H,W,C] (C multiple of 16)
//  * weights: repacked once/launch into per-lane WMMA-B fragment order (bf16)
//  * A tiles staged straight into LDS via GLOBAL_LOAD_ASYNC_TO_LDS_B128
//    (ASYNCcnt / s_wait_asynccnt), shared by up to 4 waves per block
//  * staging addresses hoisted: per-thread items precomputed once, neighbor
//    index once per tap, inner chunk loop is add + async issue only
// ---------------------------------------------------------------------------

typedef __bf16 bf16_t;
typedef __attribute__((ext_vector_type(16))) __bf16 v16bf;
typedef __attribute__((ext_vector_type(8)))  __bf16 v8bf;
typedef __attribute__((ext_vector_type(8)))  float  v8f;

#define BN_EPS 1e-5f
#define AS_STRIDE 40   // 16 rows x 40 halfwords: conflict-free row access

__device__ __forceinline__ v8f wmma_bf16(v16bf a, v16bf b, v8f c) {
  // (neg_a, A, neg_b, B, c_mod, C, reuse_a, reuse_b)
  return __builtin_amdgcn_wmma_f32_16x16x32_bf16(false, a, false, b,
                                                 (short)0, c, false, false);
}

__device__ __forceinline__ v8bf zero8() {
  v8bf z;
#pragma unroll
  for (int q = 0; q < 8; ++q) z[q] = (__bf16)0.0f;
  return z;
}

// Per-lane async copy of 16 bytes global -> LDS (CDNA5 Tensor/async path).
// LDS dest address = low 32 bits of the generic shared pointer (LDS aperture
// keeps the byte offset in addr[31:0]). Invalid runs zero-fill via ds_store.
__device__ __forceinline__ void async_stage_b128(const bf16_t* gsrc,
                                                 __bf16* ldst, bool valid) {
  if (valid) {
    unsigned lds = (unsigned)(uintptr_t)ldst;
    unsigned long long ga = (unsigned long long)(uintptr_t)gsrc;
    asm volatile("global_load_async_to_lds_b128 %0, %1, off"
                 :
                 : "v"(lds), "v"(ga)
                 : "memory");
  } else {
    *reinterpret_cast<v8bf*>(ldst) = zero8();
  }
}

__device__ __forceinline__ void wait_async_then_barrier() {
  asm volatile("s_wait_asynccnt 0x0" ::: "memory");
  __syncthreads();
}

// Build A fragment from LDS tile (two ds_load_b128) and B fragment from the
// repacked weight stream (two global_load_b128, contiguous per lane).
__device__ __forceinline__ void frags_from_lds_pack(
    const __bf16* __restrict__ As, const bf16_t* __restrict__ bbase, int idx,
    int half, v16bf& a, v16bf& b) {
  const v8bf* a0 = reinterpret_cast<const v8bf*>(As + idx * AS_STRIDE + half * 8);
  const v8bf* a1 =
      reinterpret_cast<const v8bf*>(As + idx * AS_STRIDE + 16 + half * 8);
  v8bf lo = *a0, hi = *a1;
  const v8bf* bp = reinterpret_cast<const v8bf*>(bbase + half * 16);
  v8bf b0 = bp[0], b1 = bp[1];
#pragma unroll
  for (int j = 0; j < 8; ++j) {
    a[j] = lo[j];
    a[8 + j] = hi[j];
    b[j] = b0[j];
    b[8 + j] = b1[j];
  }
}

// Fused BN(eval)+ELU+mask epilogue. D layout: lane -> N = lane&15,
// VGPR r -> M = r + 8*(lane>>4).
__device__ __forceinline__ void bn_elu_mask_store(
    v8f acc, bf16_t* __restrict__ out, const unsigned char* __restrict__ mask,
    const float* __restrict__ gam, const float* __restrict__ bet,
    const float* __restrict__ mea, const float* __restrict__ var, int co0,
    int vox0, int C_out, int nvox, int lane) {
  const int idx = lane & 15;
  const int half = lane >> 4;
  const int co = co0 + idx;
  const float s = gam[co] * rsqrtf(var[co] + BN_EPS);
  const float bo = bet[co] - mea[co] * s;
#pragma unroll
  for (int r = 0; r < 8; ++r) {
    int v = vox0 + r + 8 * half;
    if (v < nvox) {
      float y = acc[r] * s + bo;
      y = (y > 0.0f) ? y : expm1f(y);  // ELU(alpha=1)
      if (!mask[v]) y = 0.0f;
      out[(size_t)v * C_out + co] = (__bf16)y;
    }
  }
}

// Per-thread staging descriptor: up to 2 (m,run) slots of the 16x32 A tile.
struct StageItem {
  int m, run;     // tile row / 8-channel run
  int wq, hq, dq; // voxel coords (output grid)
  int base;       // voxel linear index (output grid)
  bool v;         // voxel in range
};

__device__ __forceinline__ StageItem make_item(int s, int vox0, int S,
                                               int nvox) {
  StageItem it;
  it.m = s >> 2;
  it.run = s & 3;
  int vox = vox0 + it.m;
  it.v = vox < nvox;
  int vv = it.v ? vox : 0;
  it.wq = vv % S;
  it.hq = (vv / S) % S;
  it.dq = vv / (S * S);
  it.base = (it.dq * S + it.hq) * S + it.wq;
  return it;
}

// --------------------------- weight repack ---------------------------------
// dst[(g*Cout + co)*32 + slot] = bf16 weight for K-index kpat(slot);
// slot: half = slot>>4, j = slot&15, kpat = j + 8*half + (j>=8 ? 8 : 0)
// normal mode: g = tap*nchunk + chunk, K enumerates input channels
// kIsTap mode (enc0): g = 0, K enumerates the 27 taps (Cin==1), zero-padded.
__global__ void k_repack(const float* __restrict__ src,
                         bf16_t* __restrict__ dst, int Cout, int Cin,
                         int ktaps, int nchunk, int kIsTap, long long total) {
  long long i = blockIdx.x * (long long)blockDim.x + threadIdx.x;
  if (i >= total) return;
  int slot = (int)(i & 31);
  long long rest = i >> 5;
  int co = (int)(rest % Cout);
  int g = (int)(rest / Cout);
  int half = slot >> 4, j = slot & 15;
  int kpat = j + half * 8 + ((j >= 8) ? 8 : 0);
  float v = 0.0f;
  if (kIsTap) {
    if (kpat < ktaps) v = src[(size_t)co * ktaps + kpat];
  } else {
    int tap = g / nchunk, chunk = g - tap * nchunk;
    int c = chunk * 32 + kpat;
    if (c < Cin) v = src[((size_t)co * Cin + c) * ktaps + tap];
  }
  dst[((size_t)g * Cout + co) * 32 + slot] = (__bf16)v;
}

// --------------------------- convolution kernels ---------------------------

// Submanifold 3x3x3 s1 p1, fused BN+ELU+mask. blockDim = 32*WPB, WPB<=4.
__global__ void k_subconv3(const bf16_t* __restrict__ in,
                           bf16_t* __restrict__ out,
                           const unsigned char* __restrict__ mask,
                           const bf16_t* __restrict__ pack,
                           const float* __restrict__ gam,
                           const float* __restrict__ bet,
                           const float* __restrict__ mea,
                           const float* __restrict__ var, int C_in, int C_out,
                           int S, int nvox) {
  __shared__ __bf16 As[16 * AS_STRIDE];
  const int tid = threadIdx.x;
  const int lane = tid & 31, wave = tid >> 5;
  const int idx = lane & 15, half = lane >> 4;
  const int WPB = blockDim.x >> 5;
  const int vox0 = blockIdx.x * 16;
  const int co0 = (blockIdx.y * WPB + wave) * 16;
  const int nchunk = (C_in + 31) >> 5;

  const int s0 = tid, s1 = tid + blockDim.x;
  const bool has0 = s0 < 64, has1 = s1 < 64;
  StageItem it0 = make_item(has0 ? s0 : 0, vox0, S, nvox);
  StageItem it1 = make_item(has1 ? s1 : 0, vox0, S, nvox);

  v8f acc = {};
  for (int t = 0; t < 27; ++t) {
    const int dz = t / 9 - 1, dy = (t / 3) % 3 - 1, dx = t % 3 - 1;
    const int offt = (dz * S + dy) * S + dx;  // wave-uniform neighbor delta
    // neighbor validity + base address, once per tap per item
    bool nv0 = has0 && it0.v && (unsigned)(it0.dq + dz) < (unsigned)S &&
               (unsigned)(it0.hq + dy) < (unsigned)S &&
               (unsigned)(it0.wq + dx) < (unsigned)S;
    bool nv1 = has1 && it1.v && (unsigned)(it1.dq + dz) < (unsigned)S &&
               (unsigned)(it1.hq + dy) < (unsigned)S &&
               (unsigned)(it1.wq + dx) < (unsigned)S;
    size_t nb0 = (size_t)(it0.base + offt) * C_in;
    size_t nb1 = (size_t)(it1.base + offt) * C_in;
    for (int ch = 0; ch < nchunk; ++ch) {
      __syncthreads();
      if (has0) {
        int cOff = (ch << 5) + it0.run * 8;
        bool rv = nv0 && (cOff < C_in);
        async_stage_b128(in + (rv ? nb0 + cOff : 0),
                         As + it0.m * AS_STRIDE + it0.run * 8, rv);
      }
      if (has1) {
        int cOff = (ch << 5) + it1.run * 8;
        bool rv = nv1 && (cOff < C_in);
        async_stage_b128(in + (rv ? nb1 + cOff : 0),
                         As + it1.m * AS_STRIDE + it1.run * 8, rv);
      }
      wait_async_then_barrier();
      const bf16_t* bbase =
          pack + (((size_t)t * nchunk + ch) * C_out + (co0 + idx)) * 32;
      __builtin_prefetch(bbase + (size_t)C_out * 32, 0, 1);
      v16bf a, b;
      frags_from_lds_pack(As, bbase, idx, half, a, b);
      acc = wmma_bf16(a, b, acc);
    }
  }
  bn_elu_mask_store(acc, out, mask, gam, bet, mea, var, co0, vox0, C_out, nvox,
                    lane);
}

// 2x2x2 stride-2 downsampling conv. S = output side (input side = 2S).
__global__ void k_downconv(const bf16_t* __restrict__ in,
                           bf16_t* __restrict__ out,
                           const unsigned char* __restrict__ mask,
                           const bf16_t* __restrict__ pack,
                           const float* __restrict__ gam,
                           const float* __restrict__ bet,
                           const float* __restrict__ mea,
                           const float* __restrict__ var, int C_in, int C_out,
                           int S, int nvox) {
  __shared__ __bf16 As[16 * AS_STRIDE];
  const int tid = threadIdx.x;
  const int lane = tid & 31, wave = tid >> 5;
  const int idx = lane & 15, half = lane >> 4;
  const int WPB = blockDim.x >> 5;
  const int vox0 = blockIdx.x * 16;
  const int co0 = (blockIdx.y * WPB + wave) * 16;
  const int nchunk = (C_in + 31) >> 5;
  const int Si = 2 * S;

  const int s0 = tid, s1 = tid + blockDim.x;
  const bool has0 = s0 < 64, has1 = s1 < 64;
  StageItem it0 = make_item(has0 ? s0 : 0, vox0, S, nvox);
  StageItem it1 = make_item(has1 ? s1 : 0, vox0, S, nvox);
  // base voxel index in the (2S)^3 input grid
  const int ib0 = ((2 * it0.dq) * Si + 2 * it0.hq) * Si + 2 * it0.wq;
  const int ib1 = ((2 * it1.dq) * Si + 2 * it1.hq) * Si + 2 * it1.wq;

  v8f acc = {};
  for (int t = 0; t < 8; ++t) {
    const int offt = ((((t >> 2) & 1) * Si + ((t >> 1) & 1)) * Si) + (t & 1);
    size_t nb0 = (size_t)(ib0 + offt) * C_in;
    size_t nb1 = (size_t)(ib1 + offt) * C_in;
    for (int ch = 0; ch < nchunk; ++ch) {
      __syncthreads();
      if (has0) {
        int cOff = (ch << 5) + it0.run * 8;
        bool rv = it0.v && (cOff < C_in);
        async_stage_b128(in + (rv ? nb0 + cOff : 0),
                         As + it0.m * AS_STRIDE + it0.run * 8, rv);
      }
      if (has1) {
        int cOff = (ch << 5) + it1.run * 8;
        bool rv = it1.v && (cOff < C_in);
        async_stage_b128(in + (rv ? nb1 + cOff : 0),
                         As + it1.m * AS_STRIDE + it1.run * 8, rv);
      }
      wait_async_then_barrier();
      const bf16_t* bbase =
          pack + (((size_t)t * nchunk + ch) * C_out + (co0 + idx)) * 32;
      __builtin_prefetch(bbase + (size_t)C_out * 32, 0, 1);
      v16bf a, b;
      frags_from_lds_pack(As, bbase, idx, half, a, b);
      acc = wmma_bf16(a, b, acc);
    }
  }
  bn_elu_mask_store(acc, out, mask, gam, bet, mea, var, co0, vox0, C_out, nvox,
                    lane);
}

// Transpose conv stride 2 (gather form, parity-checked taps).
// kk=2 (VALID, padlo=0) or kk=4 (SAME, padlo=1). Sout = 2*Sin.
__global__ void k_upconv(const bf16_t* __restrict__ in,
                         bf16_t* __restrict__ out,
                         const unsigned char* __restrict__ mask,
                         const bf16_t* __restrict__ pack,
                         const float* __restrict__ gam,
                         const float* __restrict__ bet,
                         const float* __restrict__ mea,
                         const float* __restrict__ var, int C_in, int C_out,
                         int Sin, int Sout, int kk, int padlo, int nvox) {
  __shared__ __bf16 As[16 * AS_STRIDE];
  const int tid = threadIdx.x;
  const int lane = tid & 31, wave = tid >> 5;
  const int idx = lane & 15, half = lane >> 4;
  const int WPB = blockDim.x >> 5;
  const int vox0 = blockIdx.x * 16;
  const int co0 = (blockIdx.y * WPB + wave) * 16;
  const int nchunk = (C_in + 31) >> 5;
  const int k3 = kk * kk * kk;

  const int s0 = tid, s1 = tid + blockDim.x;
  const bool has0 = s0 < 64, has1 = s1 < 64;
  StageItem it0 = make_item(has0 ? s0 : 0, vox0, Sout, nvox);
  StageItem it1 = make_item(has1 ? s1 : 0, vox0, Sout, nvox);

  v8f acc = {};
  for (int t = 0; t < k3; ++t) {
    const int tz = t / (kk * kk), ty = (t / kk) % kk, tx = t % kk;
    // per-item source voxel (parity-checked fractional stride), once per tap
    int ad0 = it0.dq + padlo - tz, ah0 = it0.hq + padlo - ty,
        aw0 = it0.wq + padlo - tx;
    bool nv0 = has0 && it0.v && ad0 >= 0 && ah0 >= 0 && aw0 >= 0 &&
               !(ad0 & 1) && !(ah0 & 1) && !(aw0 & 1) && (ad0 >> 1) < Sin &&
               (ah0 >> 1) < Sin && (aw0 >> 1) < Sin;
    int ad1 = it1.dq + padlo - tz, ah1 = it1.hq + padlo - ty,
        aw1 = it1.wq + padlo - tx;
    bool nv1 = has1 && it1.v && ad1 >= 0 && ah1 >= 0 && aw1 >= 0 &&
               !(ad1 & 1) && !(ah1 & 1) && !(aw1 & 1) && (ad1 >> 1) < Sin &&
               (ah1 >> 1) < Sin && (aw1 >> 1) < Sin;
    size_t nb0 =
        (size_t)((((ad0 >> 1) * Sin + (ah0 >> 1)) * Sin) + (aw0 >> 1)) * C_in;
    size_t nb1 =
        (size_t)((((ad1 >> 1) * Sin + (ah1 >> 1)) * Sin) + (aw1 >> 1)) * C_in;
    for (int ch = 0; ch < nchunk; ++ch) {
      __syncthreads();
      if (has0) {
        int cOff = (ch << 5) + it0.run * 8;
        bool rv = nv0 && (cOff < C_in);
        async_stage_b128(in + (rv ? nb0 + cOff : 0),
                         As + it0.m * AS_STRIDE + it0.run * 8, rv);
      }
      if (has1) {
        int cOff = (ch << 5) + it1.run * 8;
        bool rv = nv1 && (cOff < C_in);
        async_stage_b128(in + (rv ? nb1 + cOff : 0),
                         As + it1.m * AS_STRIDE + it1.run * 8, rv);
      }
      wait_async_then_barrier();
      const bf16_t* bbase =
          pack + (((size_t)t * nchunk + ch) * C_out + (co0 + idx)) * 32;
      __builtin_prefetch(bbase + (size_t)C_out * 32, 0, 1);
      v16bf a, b;
      frags_from_lds_pack(As, bbase, idx, half, a, b);
      acc = wmma_bf16(a, b, acc);
    }
  }
  bn_elu_mask_store(acc, out, mask, gam, bet, mea, var, co0, vox0, C_out, nvox,
                    lane);
}

// enc0: Cin=1, 3x3x3 -> single K=32 WMMA over the (zero-padded) 27 taps.
__global__ __launch_bounds__(32) void k_conv0(
    const bf16_t* __restrict__ xb, bf16_t* __restrict__ out,
    const unsigned char* __restrict__ mask, const bf16_t* __restrict__ pack,
    const float* __restrict__ gam, const float* __restrict__ bet,
    const float* __restrict__ mea, const float* __restrict__ var, int S,
    int nvox) {
  const int lane = threadIdx.x;
  const int idx = lane & 15, half = lane >> 4;
  const int vox0 = blockIdx.x * 16;
  const int vox = vox0 + idx;
  const bool vvalid = vox < nvox;
  const int vv = vvalid ? vox : 0;
  const int wq = vv % S, hq = (vv / S) % S, dq = vv / (S * S);
  v16bf a;
#pragma unroll
  for (int j = 0; j < 16; ++j) {
    int k = j + half * 8 + ((j >= 8) ? 8 : 0);
    bool tv = vvalid && (k < 27);
    int kc = tv ? k : 0;
    int nd = dq + kc / 9 - 1, nh = hq + (kc / 3) % 3 - 1, nw = wq + kc % 3 - 1;
    tv = tv && (unsigned)nd < (unsigned)S && (unsigned)nh < (unsigned)S &&
         (unsigned)nw < (unsigned)S;
    size_t lin = tv ? (size_t)(((size_t)nd * S + nh) * S + nw) : 0;
    __bf16 val = xb[lin];  // always safe
    a[j] = tv ? val : (__bf16)0.0f;
  }
  v16bf b;
  {
    const v8bf* bp =
        reinterpret_cast<const v8bf*>(pack + (size_t)idx * 32 + half * 16);
    v8bf b0 = bp[0], b1 = bp[1];
#pragma unroll
    for (int j = 0; j < 8; ++j) {
      b[j] = b0[j];
      b[8 + j] = b1[j];
    }
  }
  v8f acc = {};
  acc = wmma_bf16(a, b, acc);
  bn_elu_mask_store(acc, out, mask, gam, bet, mea, var, 0, vox0, 16, nvox,
                    lane);
}

// ---------------------------- elementwise kernels --------------------------

__global__ void k_mask_input(const float* __restrict__ x,
                             const unsigned char* __restrict__ occ,
                             bf16_t* __restrict__ xb,
                             unsigned char* __restrict__ m0, int n) {
  int i = blockIdx.x * blockDim.x + threadIdx.x;
  if (i >= n) return;
  unsigned char o = occ[i] ? 1 : 0;
  m0[i] = o;
  xb[i] = o ? (__bf16)x[i] : (__bf16)0.0f;
}

__global__ void k_down_mask(const unsigned char* __restrict__ m,
                            unsigned char* __restrict__ m2, int S, int n) {
  int i = blockIdx.x * blockDim.x + threadIdx.x;
  if (i >= n) return;
  int w = i % S, h = (i / S) % S, d = i / (S * S);
  int Si = 2 * S;
  unsigned char r = 0;
  for (int dz = 0; dz < 2; ++dz)
    for (int dy = 0; dy < 2; ++dy)
      for (int dx = 0; dx < 2; ++dx)
        r |= m[((size_t)(2 * d + dz) * Si + (2 * h + dy)) * Si + (2 * w + dx)];
  m2[i] = r ? 1 : 0;
}

__global__ void k_up_mask2(const unsigned char* __restrict__ m,
                           unsigned char* __restrict__ m2, int S, int n) {
  int i = blockIdx.x * blockDim.x + threadIdx.x;
  if (i >= n) return;
  int w = i % S, h = (i / S) % S, d = i / (S * S);
  int Si = S / 2;
  m2[i] = m[((size_t)(d >> 1) * Si + (h >> 1)) * Si + (w >> 1)];
}

__global__ void k_up_mask4(const unsigned char* __restrict__ m,
                           unsigned char* __restrict__ m2, int S, int n) {
  int i = blockIdx.x * blockDim.x + threadIdx.x;
  if (i >= n) return;
  int w = i % S, h = (i / S) % S, d = i / (S * S);
  int Si = S / 2;
  unsigned char r = 0;
  for (int t = 0; t < 64; ++t) {
    int tz = t / 16, ty = (t / 4) % 4, tx = t % 4;
    int ad = d + 1 - tz, ah = h + 1 - ty, aw = w + 1 - tx;
    if (ad < 0 || ah < 0 || aw < 0 || (ad & 1) || (ah & 1) || (aw & 1))
      continue;
    int qd = ad >> 1, qh = ah >> 1, qw = aw >> 1;
    if (qd >= Si || qh >= Si || qw >= Si) continue;
    r |= m[((size_t)qd * Si + qh) * Si + qw];
  }
  m2[i] = r ? 1 : 0;
}

__global__ void k_union(const unsigned char* __restrict__ a,
                        const unsigned char* __restrict__ b,
                        unsigned char* __restrict__ o, int n) {
  int i = blockIdx.x * blockDim.x + threadIdx.x;
  if (i >= n) return;
  o[i] = (a[i] | b[i]) ? 1 : 0;
}

__global__ void k_add(bf16_t* __restrict__ d, const bf16_t* __restrict__ f,
                      long long n) {
  long long i = blockIdx.x * (long long)blockDim.x + threadIdx.x;
  if (i >= n) return;
  d[i] = (__bf16)((float)d[i] + (float)f[i]);
}

__global__ void k_cls(const bf16_t* __restrict__ in,
                      const float* __restrict__ w, const float* __restrict__ b,
                      const unsigned char* __restrict__ m,
                      float* __restrict__ c, int C, int n) {
  int v = blockIdx.x * blockDim.x + threadIdx.x;
  if (v >= n) return;
  const bf16_t* p = in + (size_t)v * C;
  float s = b[0];
  for (int ch = 0; ch < C; ch += 8) {
    v8bf x = *reinterpret_cast<const v8bf*>(p + ch);
#pragma unroll
    for (int q = 0; q < 8; ++q) s += (float)x[q] * w[ch + q];
  }
  c[v] = m[v] ? s : 0.0f;
}

__global__ void k_set_int(int* p, int v) {
  if (threadIdx.x == 0 && blockIdx.x == 0) *p = v;
}

__global__ void k_anypos(const float* __restrict__ c,
                         const unsigned char* __restrict__ m, int* flag,
                         int n) {
  int v = blockIdx.x * blockDim.x + threadIdx.x;
  if (v >= n) return;
  if (m[v] && c[v] > 0.0f) atomicOr(flag, 1);
}

__global__ void k_newmask(const float* __restrict__ c,
                          const unsigned char* __restrict__ md,
                          const int* __restrict__ flag,
                          unsigned char* __restrict__ nm, int n) {
  int v = blockIdx.x * blockDim.x + threadIdx.x;
  if (v >= n) return;
  unsigned char keep = (md[v] && c[v] > 0.0f) ? 1 : 0;
  nm[v] = (*flag) ? keep : md[v];
}

__global__ void k_maskfeat(bf16_t* __restrict__ d,
                           const unsigned char* __restrict__ m, int C,
                           long long n) {
  long long i = blockIdx.x * (long long)blockDim.x + threadIdx.x;
  if (i >= n) return;
  int v = (int)(i / C);
  if (!m[v]) d[i] = (__bf16)0.0f;
}

__global__ void k_final(const float* __restrict__ c,
                        const unsigned char* __restrict__ md,
                        const int* __restrict__ flag, float* __restrict__ out,
                        int n) {
  int v = blockIdx.x * blockDim.x + threadIdx.x;
  if (v >= n) return;
  bool keep = md[v] && c[v] > 0.0f;
  bool nm = (*flag) ? keep : (md[v] != 0);
  out[v] = nm ? c[v] : 0.0f;
}

// ------------------------------- host driver -------------------------------
//
// Input flattening (setup_inputs() insertion order, depth-first):
//   [0] x (128^3 f32)                [1] occ (128^3 bool/u8)
//   [2] enc0.conv                    [3..6] enc0.bn {gamma,beta,mean,var}
//   enc[i] (i=0..5) @ 7+10i : down, bn1{g,b,m,v}, conv, bn2{g,b,m,v}
//   dec[j] (j=0..5) @ 67+10j: up,   bn1{g,b,m,v}, conv, bn2{g,b,m,v}
//   cls[j] (j=0..5) @ 127+2j: w (co f32), b (1 f32)

static const int CHN[7]     = {16, 32, 64, 128, 256, 512, 1024};
static const int DEC_IN[6]  = {1024, 512, 256, 128, 64, 32};
static const int DEC_OUT[6] = {512, 256, 128, 64, 32, 16};
static const int DEC_KK[6]  = {4, 2, 2, 2, 2, 2};

extern "C" void kernel_launch(void* const* d_in, const int* in_sizes, int n_in,
                              void* d_out, int out_size, void* d_ws,
                              size_t ws_size, hipStream_t stream) {
  (void)in_sizes; (void)n_in; (void)out_size; (void)ws_size;

  const float* x = (const float*)d_in[0];
  const unsigned char* occ = (const unsigned char*)d_in[1];
  auto W = [&](int i) { return (const float*)d_in[i]; };
  auto encB = [](int i) { return 7 + i * 10; };
  auto decB = [](int j) { return 67 + j * 10; };
  auto clsB = [](int j) { return 127 + j * 2; };

  char* wsb = (char*)d_ws;
  size_t off = 0;
  auto alloc = [&](size_t bytes) -> void* {
    off = (off + 255) & ~(size_t)255;
    void* p = wsb + off;
    off += bytes;
    return p;
  };

  const int EB = 256;
  auto eg = [&](long long n) { return dim3((unsigned)((n + EB - 1) / EB)); };

  // --- repack all conv weights into WMMA-B fragment order (bf16) ---
  auto repack = [&](const float* src, int Cout, int Cin, int ktaps,
                    int kIsTap) -> bf16_t* {
    int nchunk = kIsTap ? 1 : (Cin + 31) >> 5;
    long long groups = kIsTap ? 1 : (long long)ktaps * nchunk;
    long long total = groups * Cout * 32;
    bf16_t* p = (bf16_t*)alloc((size_t)total * sizeof(bf16_t));
    k_repack<<<eg(total), EB, 0, stream>>>(src, p, Cout, Cin, ktaps, nchunk,
                                           kIsTap, total);
    return p;
  };

  bf16_t* p_enc0 = repack(W(2), 16, 1, 27, 1);
  bf16_t *p_down[6], *p_econv[6], *p_up[6], *p_dconv[6];
  for (int i = 0; i < 6; ++i) {
    int ci = CHN[i], co = CHN[i + 1];
    p_down[i] = repack(W(encB(i)), co, ci, 8, 0);
    p_econv[i] = repack(W(encB(i) + 5), co, co, 27, 0);
  }
  for (int j = 0; j < 6; ++j) {
    int ci = DEC_IN[j], co = DEC_OUT[j], kk = DEC_KK[j];
    p_up[j] = repack(W(decB(j)), co, ci, kk * kk * kk, 0);
    p_dconv[j] = repack(W(decB(j) + 5), co, co, 27, 0);
  }

  int side[7], nv[7];
  for (int i = 0; i < 7; ++i) {
    side[i] = 128 >> i;
    nv[i] = side[i] * side[i] * side[i];
  }

  unsigned char* ms[7];
  bf16_t* f[7];
  for (int i = 0; i < 7; ++i) {
    ms[i] = (unsigned char*)alloc(nv[i]);
    f[i] = (bf16_t*)alloc((size_t)nv[i] * CHN[i] * sizeof(bf16_t));
  }
  bf16_t* xb = (bf16_t*)alloc((size_t)nv[0] * sizeof(bf16_t));
  // shared conv-intermediate scratch (max: decoder stage 5 = 2M vox x 16 ch)
  bf16_t* tmp = (bf16_t*)alloc((size_t)nv[0] * 16 * sizeof(bf16_t));

  auto convGrid = [&](int nvox, int Cout, int& wpb) {
    wpb = (Cout >= 64) ? 4 : (Cout / 16);
    return dim3((nvox + 15) / 16, Cout / (16 * wpb));
  };

  // ---- input masking + enc0 ----
  k_mask_input<<<eg(nv[0]), EB, 0, stream>>>(x, occ, xb, ms[0], nv[0]);
  k_conv0<<<dim3((nv[0] + 15) / 16), 32, 0, stream>>>(
      xb, f[0], ms[0], p_enc0, W(3), W(4), W(5), W(6), side[0], nv[0]);

  // ---- encoder ----
  for (int i = 0; i < 6; ++i) {
    int co = CHN[i + 1];
    int so = side[i + 1], no = nv[i + 1];
    int b = encB(i);
    k_down_mask<<<eg(no), EB, 0, stream>>>(ms[i], ms[i + 1], so, no);
    int wpb;
    dim3 g = convGrid(no, co, wpb);
    k_downconv<<<g, 32 * wpb, 0, stream>>>(f[i], tmp, ms[i + 1], p_down[i],
                                           W(b + 1), W(b + 2), W(b + 3),
                                           W(b + 4), CHN[i], co, so, no);
    k_subconv3<<<g, 32 * wpb, 0, stream>>>(tmp, f[i + 1], ms[i + 1], p_econv[i],
                                           W(b + 6), W(b + 7), W(b + 8),
                                           W(b + 9), co, co, so, no);
  }

  // ---- decoder ----
  bf16_t* dcur = f[6];
  unsigned char* mdcur = ms[6];
  for (int j = 0; j < 6; ++j) {
    int sin = 128 >> (6 - j), sout = sin * 2;
    int no = sout * sout * sout;
    int ci = DEC_IN[j], co = DEC_OUT[j], kk = DEC_KK[j];
    int b = decB(j);

    unsigned char* m2 = (unsigned char*)alloc(no);
    if (kk == 4)
      k_up_mask4<<<eg(no), EB, 0, stream>>>(mdcur, m2, sout, no);
    else
      k_up_mask2<<<eg(no), EB, 0, stream>>>(mdcur, m2, sout, no);

    int wpb;
    dim3 g = convGrid(no, co, wpb);
    k_upconv<<<g, 32 * wpb, 0, stream>>>(dcur, tmp, m2, p_up[j], W(b + 1),
                                         W(b + 2), W(b + 3), W(b + 4), ci, co,
                                         sin, sout, kk, (kk == 4) ? 1 : 0, no);

    bf16_t* d2 = (bf16_t*)alloc((size_t)no * co * sizeof(bf16_t));
    k_subconv3<<<g, 32 * wpb, 0, stream>>>(tmp, d2, m2, p_dconv[j], W(b + 6),
                                           W(b + 7), W(b + 8), W(b + 9), co, co,
                                           sout, no);

    unsigned char* mdn = (unsigned char*)alloc(no);
    k_union<<<eg(no), EB, 0, stream>>>(m2, ms[5 - j], mdn, no);
    k_add<<<eg((long long)no * co), EB, 0, stream>>>(d2, f[5 - j],
                                                     (long long)no * co);

    float* cbuf = (float*)alloc((size_t)no * sizeof(float));
    k_cls<<<eg(no), EB, 0, stream>>>(d2, W(clsB(j)), W(clsB(j) + 1), mdn, cbuf,
                                     co, no);

    int* flag = (int*)alloc(sizeof(int));
    k_set_int<<<1, 1, 0, stream>>>(flag, 0);
    k_anypos<<<eg(no), EB, 0, stream>>>(cbuf, mdn, flag, no);

    if (j < 5) {
      unsigned char* nm = (unsigned char*)alloc(no);
      k_newmask<<<eg(no), EB, 0, stream>>>(cbuf, mdn, flag, nm, no);
      k_maskfeat<<<eg((long long)no * co), EB, 0, stream>>>(
          d2, nm, co, (long long)no * co);
      dcur = d2;
      mdcur = nm;
    } else {
      k_final<<<eg(no), EB, 0, stream>>>(cbuf, mdn, flag, (float*)d_out, no);
    }
  }
}
